// Attention_InfLoRA_764504179514
// MI455X (gfx1250) — compile-verified, matrix operand order checked by reading, not run on hardware
//
#include <hip/hip_runtime.h>
#include <hip/hip_bf16.h>
#include <cstdint>

// ---------------------------------------------------------------------------
// Types for gfx1250 WMMA (wave32): v_wmma_f32_16x16x32_bf16
// ---------------------------------------------------------------------------
typedef __attribute__((ext_vector_type(16))) __bf16 bf16x16;
typedef __attribute__((ext_vector_type(8)))  __bf16 bf16x8;
typedef __attribute__((ext_vector_type(4)))  __bf16 bf16x4;
typedef __attribute__((ext_vector_type(8)))  float  v8f;
typedef __attribute__((__vector_size__(4 * sizeof(int)))) int v4i;

__device__ __forceinline__ v8f wmma_bf16(bf16x16 a, bf16x16 b, v8f c) {
    return __builtin_amdgcn_wmma_f32_16x16x32_bf16(false, a, false, b,
                                                   (short)0, c, false, false);
}

// Load one 16x32 bf16 fragment (A-layout; B uses the same layout with lane=N).
// `p` points at row_base + 8*half; elems 0..7 <- K[0..7], 8..15 <- K[16..23].
__device__ __forceinline__ bf16x16 frag_ld(const __bf16* p) {
    bf16x8 lo = *(const bf16x8*)(p);
    bf16x8 hi = *(const bf16x8*)(p + 16);
    bf16x16 f;
#pragma unroll
    for (int i = 0; i < 8; ++i) { f[i] = lo[i]; f[i + 8] = hi[i]; }
    return f;
}

// ---------------------------------------------------------------------------
// gfx1250 async global->LDS (ASYNCcnt-tracked), guarded so we always compile.
// Builtin exists in this toolchain; params are (v4i AS1*, v4i AS3*, Ii, Ii).
// ---------------------------------------------------------------------------
#if defined(__has_builtin)
#if __has_builtin(__builtin_amdgcn_global_load_async_to_lds_b128)
#define USE_ASYNC_LDS 1
#endif
#endif

#define AS1 __attribute__((address_space(1)))
#define AS3 __attribute__((address_space(3)))

__device__ __forceinline__ void async_copy16(const __bf16* g, __bf16* l) {
#if defined(USE_ASYNC_LDS)
    __builtin_amdgcn_global_load_async_to_lds_b128(
        (AS1 v4i*)(uintptr_t)g,
        (AS3 v4i*)(uint32_t)(uintptr_t)l,   // LDS offset = low 32 bits of flat
        0, 0);
#else
    *(bf16x8*)l = *(const bf16x8*)g;
#endif
}

__device__ __forceinline__ void wait_async_lds() {
#if defined(USE_ASYNC_LDS)
#if __has_builtin(__builtin_amdgcn_s_wait_asynccnt)
    __builtin_amdgcn_s_wait_asynccnt(0);
#else
    asm volatile("s_wait_asynccnt 0" ::: "memory");
#endif
#endif
}

// ---------------------------------------------------------------------------
// Problem constants
// ---------------------------------------------------------------------------
#define CDIM 1024
#define NSEQ 1024
#define BDIM 16
#define HEADS 16
#define HDIM 64
#define RANK 64
#define MROWS (BDIM * NSEQ)     // 16384
#define LDT 72                  // attention LDS row stride (bf16 elems)
#define LDG 40                  // GEMM LDS row stride for 32-wide K tiles

// ---------------------------------------------------------------------------
// Kernel 1: W_eff[3C][C] = W_qkv + [0 ; sum_t B_k@A_k ; sum_t B_v@A_v]  (bf16)
// ---------------------------------------------------------------------------
__global__ __launch_bounds__(256)
void build_weff(const float* __restrict__ Wqkv,
                const float* __restrict__ A_k, const float* __restrict__ B_k,
                const float* __restrict__ A_v, const float* __restrict__ B_v,
                const int* __restrict__ taskp, __bf16* __restrict__ Weff) {
    const int idx = blockIdx.x * 256 + threadIdx.x;      // 3C*C threads
    const int col = idx & (CDIM - 1);
    const int row = idx >> 10;
    const int T = *taskp;
    float acc = Wqkv[idx];
    const float* Am = nullptr; const float* Bm = nullptr; int c = 0;
    if (row >= CDIM && row < 2 * CDIM) { Am = A_k; Bm = B_k; c = row - CDIM; }
    else if (row >= 2 * CDIM)          { Am = A_v; Bm = B_v; c = row - 2 * CDIM; }
    if (Am) {
        for (int t = 0; t <= T; ++t) {
            const float* Brow = Bm + ((size_t)t * CDIM + c) * RANK;
            const float* Acol = Am + (size_t)t * RANK * CDIM + col;
#pragma unroll 8
            for (int r = 0; r < RANK; ++r)
                acc += Brow[r] * Acol[(size_t)r * CDIM];
        }
    }
    Weff[idx] = (__bf16)acc;
}

// ---------------------------------------------------------------------------
// Kernel 2: elementwise f32 -> bf16
// ---------------------------------------------------------------------------
__global__ __launch_bounds__(256)
void conv_bf16(const float* __restrict__ in, __bf16* __restrict__ out, int n) {
    int i = blockIdx.x * 256 + threadIdx.x;
    if (i < n) out[i] = (__bf16)in[i];
}

// ---------------------------------------------------------------------------
// Kernel 3: Out[M][N] = A[M][K] @ Bw[N][K]^T (+bias).  Pure bf16 WMMA GEMM.
// 256 thr (8 waves), block tile 128x128, BK=32, double-buffered LDS with
// async global->LDS copies issued one tile ahead. Wave (wm,wn) -> 32x64.
// ---------------------------------------------------------------------------
__device__ __forceinline__ void stage128x32(__bf16* lds, const __bf16* g,
                                            int gld, int tid) {
#pragma unroll
    for (int i = 0; i < 2; ++i) {
        const int c  = tid + i * 256;       // 512 chunks of 8 bf16
        const int r  = c >> 2;
        const int cc = (c & 3) << 3;
        async_copy16(g + (size_t)r * gld + cc, lds + r * LDG + cc);
    }
}

template <typename OT>
__global__ __launch_bounds__(256)
void gemm_xwt(const __bf16* __restrict__ A, const __bf16* __restrict__ Bw,
              OT* __restrict__ Out, const float* __restrict__ bias,
              int M, int N, int K) {
    __shared__ __bf16 As[2][128 * LDG];
    __shared__ __bf16 Bs[2][128 * LDG];
    const int tid  = threadIdx.x;
    const int lane = tid & 31;
    const int wave = tid >> 5;
    const int m16  = lane & 15;
    const int half = lane >> 4;
    const int wm   = wave >> 1;             // 0..3 -> 32-row slice
    const int wn   = wave & 1;              // 0..1 -> 64-col slice
    const int nBase = blockIdx.x * 128;
    const int mBase = blockIdx.y * 128;

    const __bf16* Ag = A  + (size_t)mBase * K;
    const __bf16* Bg = Bw + (size_t)nBase * K;

    v8f acc[2][4] = {};

    // prologue: stage tile 0
    stage128x32(As[0], Ag, K, tid);
    stage128x32(Bs[0], Bg, K, tid);
    wait_async_lds();
    __syncthreads();

    int buf = 0;
    for (int k0 = 0; k0 < K; k0 += 32, buf ^= 1) {
        // issue async copies for next tile into the other buffer
        if (k0 + 32 < K) {
            stage128x32(As[buf ^ 1], Ag + k0 + 32, K, tid);
            stage128x32(Bs[buf ^ 1], Bg + k0 + 32, K, tid);
        }
        // compute on current buffer
        const __bf16* Ab = As[buf];
        const __bf16* Bb = Bs[buf];
        bf16x16 af0 = frag_ld(&Ab[(wm * 32 +  0 + m16) * LDG + 8 * half]);
        bf16x16 af1 = frag_ld(&Ab[(wm * 32 + 16 + m16) * LDG + 8 * half]);
#pragma unroll
        for (int nt = 0; nt < 4; ++nt) {
            bf16x16 bfr = frag_ld(&Bb[(wn * 64 + nt * 16 + m16) * LDG + 8 * half]);
            acc[0][nt] = wmma_bf16(af0, bfr, acc[0][nt]);
            acc[1][nt] = wmma_bf16(af1, bfr, acc[1][nt]);
        }
        wait_async_lds();       // next tile landed (overlapped with WMMAs)
        __syncthreads();        // everyone done reading cur buf
    }

#pragma unroll
    for (int mt = 0; mt < 2; ++mt) {
#pragma unroll
        for (int nt = 0; nt < 4; ++nt) {
            const int col = nBase + wn * 64 + nt * 16 + m16;
            const float bv = bias ? bias[col] : 0.0f;
#pragma unroll
            for (int r = 0; r < 8; ++r) {
                const int row = mBase + wm * 32 + mt * 16 + r + 8 * half;
                Out[(size_t)row * N + col] = (OT)(acc[mt][nt][r] + bv);
            }
        }
    }
}

// ---------------------------------------------------------------------------
// Kernel 4: flash attention per (b,h). qkv bf16 [16384][3072] (q|k|v blocks).
// Block = 128 thr (4 waves), one 64-row q-block; loop over 64-row kv blocks.
// ---------------------------------------------------------------------------
__global__ __launch_bounds__(128)
void attn_kernel(const __bf16* __restrict__ qkv, __bf16* __restrict__ Ob) {
    const int bh   = blockIdx.y;
    const int b    = bh >> 4;
    const int h    = bh & 15;
    const int qBase = blockIdx.x * 64;
    const int tid  = threadIdx.x;
    const int lane = tid & 31;
    const int wave = tid >> 5;
    const int m16  = lane & 15;
    const int half = lane >> 4;

    __shared__ __bf16 Qs[64 * LDT];
    __shared__ __bf16 Ks[64 * LDT];
    __shared__ __bf16 Vts[64 * LDT];         // V transposed: [d][kv]
    __shared__ __bf16 Ps[4][16 * LDT];       // per-wave P tile (A layout src)

    const size_t rs = 3 * CDIM;
    const __bf16* qptr = qkv + ((size_t)b * NSEQ + qBase) * rs + h * HDIM;
#pragma unroll
    for (int q = tid; q < 64 * 16; q += 128) {
        const int r  = q >> 4;
        const int cq = (q & 15) << 2;
        *(bf16x4*)&Qs[r * LDT + cq] = *(const bf16x4*)(qptr + (size_t)r * rs + cq);
    }

    float mrow[8], lrow[8];
    v8f o[4] = {};
#pragma unroll
    for (int r = 0; r < 8; ++r) { mrow[r] = -1e30f; lrow[r] = 0.0f; }
    const float scale = 0.125f;              // 1/sqrt(64)

    for (int kv0 = 0; kv0 < NSEQ; kv0 += 64) {
        const __bf16* kptr = qkv + ((size_t)b * NSEQ + kv0) * rs + CDIM     + h * HDIM;
        const __bf16* vptr = qkv + ((size_t)b * NSEQ + kv0) * rs + 2 * CDIM + h * HDIM;
        __syncthreads();
#pragma unroll
        for (int q = tid; q < 64 * 16; q += 128) {
            const int r  = q >> 4;
            const int cq = (q & 15) << 2;
            *(bf16x4*)&Ks[r * LDT + cq] = *(const bf16x4*)(kptr + (size_t)r * rs + cq);
            bf16x4 vv = *(const bf16x4*)(vptr + (size_t)r * rs + cq);
            Vts[(cq + 0) * LDT + r] = vv[0];
            Vts[(cq + 1) * LDT + r] = vv[1];
            Vts[(cq + 2) * LDT + r] = vv[2];
            Vts[(cq + 3) * LDT + r] = vv[3];
        }
        __syncthreads();

        // S = Q @ K^T  (wave's 16 rows x 64 kv cols)
        v8f s[4] = {};
#pragma unroll
        for (int d0 = 0; d0 < HDIM; d0 += 32) {
            bf16x16 aq = frag_ld(&Qs[(wave * 16 + m16) * LDT + d0 + 8 * half]);
#pragma unroll
            for (int nt = 0; nt < 4; ++nt) {
                bf16x16 bk = frag_ld(&Ks[(nt * 16 + m16) * LDT + d0 + 8 * half]);
                s[nt] = wmma_bf16(aq, bk, s[nt]);
            }
        }

        // online softmax (f32), stage P into LDS in A-fragment source layout
#pragma unroll
        for (int r = 0; r < 8; ++r) {
            float v0 = s[0][r] * scale, v1 = s[1][r] * scale;
            float v2 = s[2][r] * scale, v3 = s[3][r] * scale;
            float mx = fmaxf(fmaxf(v0, v1), fmaxf(v2, v3));
#pragma unroll
            for (int off = 1; off < 16; off <<= 1)
                mx = fmaxf(mx, __shfl_xor(mx, off));
            const float mnew = fmaxf(mrow[r], mx);
            const float corr = __expf(mrow[r] - mnew);
            const float p0 = __expf(v0 - mnew), p1 = __expf(v1 - mnew);
            const float p2 = __expf(v2 - mnew), p3 = __expf(v3 - mnew);
            float ps = p0 + p1 + p2 + p3;
#pragma unroll
            for (int off = 1; off < 16; off <<= 1)
                ps += __shfl_xor(ps, off);
            lrow[r] = lrow[r] * corr + ps;
            mrow[r] = mnew;
            o[0][r] *= corr; o[1][r] *= corr; o[2][r] *= corr; o[3][r] *= corr;
            __bf16* pp = &Ps[wave][(r + 8 * half) * LDT + m16];
            pp[0]  = (__bf16)p0;
            pp[16] = (__bf16)p1;
            pp[32] = (__bf16)p2;
            pp[48] = (__bf16)p3;
        }

        // O += P @ V   (B source = V^T rows, contiguous in kv)
#pragma unroll
        for (int c0 = 0; c0 < 64; c0 += 32) {
            bf16x16 ap = frag_ld(&Ps[wave][m16 * LDT + c0 + 8 * half]);
#pragma unroll
            for (int dt = 0; dt < 4; ++dt) {
                bf16x16 bv = frag_ld(&Vts[(dt * 16 + m16) * LDT + c0 + 8 * half]);
                o[dt] = wmma_bf16(ap, bv, o[dt]);
            }
        }
    }

#pragma unroll
    for (int r = 0; r < 8; ++r) {
        const float inv = 1.0f / lrow[r];
        const size_t row = (size_t)b * NSEQ + qBase + wave * 16 + r + 8 * half;
        __bf16* op = Ob + row * CDIM + h * HDIM + m16;
        op[0]  = (__bf16)(o[0][r] * inv);
        op[16] = (__bf16)(o[1][r] * inv);
        op[32] = (__bf16)(o[2][r] * inv);
        op[48] = (__bf16)(o[3][r] * inv);
    }
}

// ---------------------------------------------------------------------------
// Launch
// ---------------------------------------------------------------------------
extern "C" void kernel_launch(void* const* d_in, const int* in_sizes, int n_in,
                              void* d_out, int out_size, void* d_ws, size_t ws_size,
                              hipStream_t stream) {
    const float* x     = (const float*)d_in[0];
    const float* Wqkv  = (const float*)d_in[1];
    const float* Wproj = (const float*)d_in[2];
    const float* bproj = (const float*)d_in[3];
    const float* A_k   = (const float*)d_in[4];
    const float* B_k   = (const float*)d_in[5];
    const float* A_v   = (const float*)d_in[6];
    const float* B_v   = (const float*)d_in[7];
    const int*   taskp = (const int*)d_in[8];

    char* ws = (char*)d_ws;
    __bf16* Weff   = (__bf16*)(ws);                          // 6 MB
    __bf16* Wprojb = (__bf16*)(ws + (size_t)6291456);        // 2 MB
    __bf16* xb     = (__bf16*)(ws + (size_t)8388608);        // 32 MB
    __bf16* qkvb   = (__bf16*)(ws + (size_t)41943040);       // 96 MB
    __bf16* Ob     = (__bf16*)(ws + (size_t)142606336);      // 32 MB

    // 1) merged effective QKV weights (LoRA folded into K/V rows), bf16
    build_weff<<<dim3((3 * CDIM * CDIM) / 256), 256, 0, stream>>>(
        Wqkv, A_k, B_k, A_v, B_v, taskp, Weff);

    // 2) W_proj -> bf16 ; x -> bf16
    conv_bf16<<<dim3((CDIM * CDIM) / 256), 256, 0, stream>>>(
        Wproj, Wprojb, CDIM * CDIM);
    conv_bf16<<<dim3((MROWS * CDIM) / 256), 256, 0, stream>>>(
        x, xb, MROWS * CDIM);

    // 3) qkv = x @ W_eff^T  (16384 x 3072 x 1024), bf16 out
    gemm_xwt<__bf16><<<dim3(3 * CDIM / 128, MROWS / 128), 256, 0, stream>>>(
        xb, Weff, qkvb, nullptr, MROWS, 3 * CDIM, CDIM);

    // 4) flash attention per (b,h)
    attn_kernel<<<dim3(NSEQ / 64, BDIM * HEADS), 128, 0, stream>>>(qkvb, Ob);

    // 5) out = O @ W_proj^T + b_proj  (16384 x 1024 x 1024), f32 out
    gemm_xwt<float><<<dim3(CDIM / 128, MROWS / 128), 256, 0, stream>>>(
        Ob, Wprojb, (float*)d_out, bproj, MROWS, CDIM, CDIM);
}
